// AdaptiveWidthMultiheadAttention_88759794139163
// MI455X (gfx1250) — matrix operand order repaired and checked by reading, not compiled, not run on hardware
//
#include <hip/hip_runtime.h>
#include <hip/hip_bf16.h>
#include <stdint.h>

// ---- problem constants (match reference) ----
#define BB   2
#define TT   256
#define DM   512
#define HH   8
#define DH   64
#define HID  128
#define GH   64
#define NMEM 32768
#define TOPK 32

typedef __attribute__((ext_vector_type(16))) _Float16 v16h;
typedef __attribute__((ext_vector_type(8)))  _Float16 v8h;
typedef __attribute__((ext_vector_type(8)))  float    v8f;

// ---- CDNA5 async global->LDS copy (per-lane 16B), tracked by ASYNCcnt ----
__device__ __forceinline__ void async_copy_b128(void* lds_dst, const void* gsrc) {
    unsigned lds = (unsigned)(unsigned long long)(uintptr_t)lds_dst; // low 32 bits = LDS offset
    asm volatile("global_load_async_to_lds_b128 %0, %1, off"
                 :: "v"(lds), "v"(gsrc) : "memory");
}
__device__ __forceinline__ void wait_asynccnt0() {
    asm volatile("s_wait_asynccnt 0x0" ::: "memory");
}

// -----------------------------------------------------------------------------
// LDS-staged WMMA GEMM: C = act(A[MxK] @ B[KxN] + bias)
// Block = 128 threads (4 waves) computes a 64x64 C tile; wave w owns rows
// [16w,16w+16) x 64 cols (4 f32 accumulators, 4 WMMAs per k-step).
// A tile staged f32 via GLOBAL_LOAD_ASYNC_TO_LDS_B128 (converted at fragment
// build, amortized over 4 WMMAs); B tile staged transposed *as f16* with
// padded stride 40 halves (bank-conflict-free b128 fragment reads, no cvt).
// -----------------------------------------------------------------------------
__global__ __launch_bounds__(128)
void gemm_wmma_f16(const float* __restrict__ A, const float* __restrict__ Bm,
                   const float* __restrict__ bias, float* __restrict__ C,
                   int M, int N, int K, int act, int out_mode)
{
    __shared__ float    A_s[64 * 32];   // [row][k] f32, 8 KB (async-staged)
    __shared__ _Float16 Bt_s[64 * 40];  // [col][k] f16, stride 40, 5 KB

    const int tid  = threadIdx.x;
    const int wv   = tid >> 5;
    const int lane = tid & 31;
    const int g = lane >> 4;
    const int l = lane & 15;
    const int m0 = blockIdx.y * 64;
    const int n0 = blockIdx.x * 64;
    (void)M;

    v8f acc0 = {}, acc1 = {}, acc2 = {}, acc3 = {};

    for (int k0 = 0; k0 < K; k0 += 32) {
        // ---- stage A tile 64x32 f32 via async global->LDS (512 x 16B) ----
        for (int i = tid; i < 512; i += 128) {
            int row = i >> 3, c4 = (i & 7) * 4;
            async_copy_b128(&A_s[row * 32 + c4],
                            A + (size_t)(m0 + row) * K + k0 + c4);
        }
        // ---- stage B tile 32x64 -> f16 transposed Bt_s[col][k] ----
        for (int i = tid; i < 512; i += 128) {
            int kr = i >> 4, c = (i & 15) * 4;
            float4 vv = *(const float4*)(Bm + (size_t)(k0 + kr) * N + n0 + c);
            Bt_s[(c + 0) * 40 + kr] = (_Float16)vv.x;
            Bt_s[(c + 1) * 40 + kr] = (_Float16)vv.y;
            Bt_s[(c + 2) * 40 + kr] = (_Float16)vv.z;
            Bt_s[(c + 3) * 40 + kr] = (_Float16)vv.w;
        }
        wait_asynccnt0();
        __syncthreads();

        // ---- A fragment (rows 16w..16w+15), contiguous K: k = 16g + e ----
        const float* ar = A_s + (16 * wv + l) * 32;
        v16h af;
#pragma unroll
        for (int e = 0; e < 16; ++e)
            af[e] = (_Float16)ar[e + 16 * g];

        // ---- 4 B fragments (2x ds_load_b128 each) / 4 WMMAs ----
#pragma unroll
        for (int j = 0; j < 4; ++j) {
            const _Float16* br = &Bt_s[(16 * j + l) * 40 + 16 * g];
            v8h blo = *(const v8h*)br;
            v8h bhi = *(const v8h*)(br + 8);
            v16h bf;
#pragma unroll
            for (int e = 0; e < 8; ++e) { bf[e] = blo[e]; bf[e + 8] = bhi[e]; }
            v8f* pa = (j == 0) ? &acc0 : (j == 1) ? &acc1 : (j == 2) ? &acc2 : &acc3;
            *pa = __builtin_amdgcn_wmma_f32_16x16x32_f16(false, af, false, bf,
                                                         (short)0, *pa, false, false);
        }
        __syncthreads();
    }

#pragma unroll
    for (int j = 0; j < 4; ++j) {
        const v8f& acc = (j == 0) ? acc0 : (j == 1) ? acc1 : (j == 2) ? acc2 : acc3;
#pragma unroll
        for (int r = 0; r < 8; ++r) {
            int row = m0 + 16 * wv + 8 * g + r;  // C/D layout: VGPR r -> M = 8g + r
            int col = n0 + 16 * j + l;
            float v = acc[r];
            if (bias) v += bias[col];
            if (act == 1) {
                float u = 0.7978845608028654f * (v + 0.044715f * v * v * v);
                v = 0.5f * v * (1.0f + tanhf(u));
            }
            if (out_mode == 1) {
                int b = row >> 8, t = row & 255;
                int h = col >> 6, d = col & 63;
                C[((b * HH + h) * TT + t) * DH + d] = v;
            } else {
                C[(size_t)row * N + col] = v;
            }
        }
    }
}

// -----------------------------------------------------------------------------
// gates = clip(sigmoid(g1 @ Wg2 + bg2), 0, 1);  g1 = gelu(x@Wg1+bg1), (512 x 64)
// -----------------------------------------------------------------------------
__global__ __launch_bounds__(64)
void gates_kernel(const float* __restrict__ g1, const float* __restrict__ Wg2,
                  const float* __restrict__ bg2, float* __restrict__ gates)
{
    __shared__ float part[64][HH];
    int row = blockIdx.x, tid = threadIdx.x;
    float gv = g1[row * GH + tid];
#pragma unroll
    for (int h = 0; h < HH; ++h) part[tid][h] = gv * Wg2[tid * HH + h];
    __syncthreads();
    for (int stride = 32; stride > 0; stride >>= 1) {
        if (tid < stride) {
#pragma unroll
            for (int h = 0; h < HH; ++h) part[tid][h] += part[tid + stride][h];
        }
        __syncthreads();
    }
    if (tid < HH) {
        float s = part[0][tid] + bg2[tid];
        s = 1.0f / (1.0f + expf(-s));
        if (s < 0.f) s = 0.f;
        if (s > 1.f) s = 1.f;
        gates[row * HH + tid] = s;
    }
}

// -----------------------------------------------------------------------------
// Memory selection: q_mean over heads, 32768 scores in dynamic LDS (128 KB,
// legal on CDNA5 320KB WGP LDS), 32 argmax passes, gather K_mem / V_mem rows.
// -----------------------------------------------------------------------------
__global__ __launch_bounds__(256)
void memselect_kernel(const float* __restrict__ qB, const float* __restrict__ mem_keys,
                      const float* __restrict__ mem_values,
                      int* __restrict__ idx_out,
                      float* __restrict__ kmemA, float* __restrict__ vmemA)
{
    extern __shared__ float sm[];
    float* scores = sm;                 // [NMEM]
    float* qm     = sm + NMEM;          // [DH]
    float* rs     = qm + DH;            // [256]
    int*   ri     = (int*)(rs + 256);   // [256]

    int row = blockIdx.x;               // b*T + t
    int b = row >> 8, t = row & 255;
    int tid = threadIdx.x;

    if (tid < DH) {
        float s = 0.f;
#pragma unroll
        for (int h = 0; h < HH; ++h)
            s += qB[((b * HH + h) * TT + t) * DH + tid];
        qm[tid] = s * (1.0f / HH);
    }
    __syncthreads();

    for (int n = tid; n < NMEM; n += 256) {
        const float4* kr = (const float4*)(mem_keys + n * DH);
        __builtin_prefetch(mem_keys + (n + 256) * DH, 0, 1);  // global_prefetch_b8
        float s = 0.f;
#pragma unroll
        for (int d4 = 0; d4 < DH / 4; ++d4) {
            float4 kv = kr[d4];
            s += qm[d4 * 4 + 0] * kv.x + qm[d4 * 4 + 1] * kv.y +
                 qm[d4 * 4 + 2] * kv.z + qm[d4 * 4 + 3] * kv.w;
        }
        scores[n] = s;
    }
    __syncthreads();

    for (int it = 0; it < TOPK; ++it) {
        float bs = -3.4e38f; int bi = 0;
        for (int n = tid; n < NMEM; n += 256) {
            float s = scores[n];
            if (s > bs) { bs = s; bi = n; }
        }
        rs[tid] = bs; ri[tid] = bi;
        __syncthreads();
        for (int stride = 128; stride > 0; stride >>= 1) {
            if (tid < stride && rs[tid + stride] > rs[tid]) {
                rs[tid] = rs[tid + stride]; ri[tid] = ri[tid + stride];
            }
            __syncthreads();
        }
        int best = ri[0];
        if (tid == 0) { idx_out[row * TOPK + it] = best; scores[best] = -3.4e38f; }
        __syncthreads();
    }

    for (int e = tid; e < TOPK * DH; e += 256) {
        int k = e >> 6, d = e & 63;
        int n = idx_out[row * TOPK + k];
        kmemA[(row * TOPK + k) * DH + d] = mem_keys[n * DH + d];
        vmemA[(row * TOPK + k) * DH + d] = mem_values[n * DH + d];
    }
}

// -----------------------------------------------------------------------------
// Fused attention per (b, h, 16-row t-tile), 128 threads (4 waves):
//  - hq tile + hk_ctx tiles async-staged to LDS, relu-additive scores (VALU)
//  - softmax over 288 (also writes f16 copy of weights, padded stride 296)
//  - w_ctx @ v via WMMA with v-tiles staged to LDS as f16 (stride 40)
//  - memory-branch accumulation, gating, store
// -----------------------------------------------------------------------------
__global__ __launch_bounds__(128)
void attn_kernel(const float* __restrict__ hq, const float* __restrict__ hkc,
                 const float* __restrict__ hkm, const float* __restrict__ vB,
                 const float* __restrict__ vmemA, const float* __restrict__ gates,
                 const float* __restrict__ Ws2, const float* __restrict__ bs2,
                 float* __restrict__ outF)
{
    __shared__ float    hq_s[16 * HID];          // 8 KB (async-staged)
    __shared__ float    hk_s[32 * HID];          // 16 KB (async-staged, per s-chunk)
    __shared__ float    w_s[16 * (TT + TOPK)];   // 16 x 288 f32, 18 KB
    __shared__ _Float16 wh_s[16 * 296];          // f16 weights, stride 296, 9.25 KB
    __shared__ _Float16 Vt_s[64 * 40];           // v-tile [col][k] f16, 5 KB
    __shared__ float    ws2_s[HID];

    const int tT = blockIdx.x;
    const int h  = blockIdx.y;
    const int b  = blockIdx.z;
    const int tid = threadIdx.x;
    const int t0 = tT * 16;
    const int bh = b * HH + h;
    const int SW = TT + TOPK;    // 288

    // stage hq rows (16 x 128 f32 = 512 x 16B) via async
    for (int i = tid; i < 512; i += 128) {
        int t = i >> 5, c4 = (i & 31) * 4;
        async_copy_b128(&hq_s[t * HID + c4],
                        hq + (size_t)(bh * TT + t0 + t) * HID + c4);
    }
    ws2_s[tid] = Ws2[tid];
    wait_asynccnt0();
    __syncthreads();
    const float b2 = bs2[0];

    // context scores in 32-row hk chunks staged via async
    for (int sc = 0; sc < TT; sc += 32) {
        for (int i = tid; i < 1024; i += 128) {
            int r = i >> 5, c4 = (i & 31) * 4;
            async_copy_b128(&hk_s[r * HID + c4],
                            hkc + (size_t)(bh * TT + sc + r) * HID + c4);
        }
        wait_asynccnt0();
        __syncthreads();
        for (int e = tid; e < 16 * 32; e += 128) {
            int t = e >> 5, s = e & 31;
            const float* hk  = hk_s + s * HID;
            const float* hqr = hq_s + t * HID;
            float acc = 0.f;
            for (int f = 0; f < HID; ++f) {
                float u = hqr[f] + hk[f];
                acc += (u > 0.f ? u : 0.f) * ws2_s[f];
            }
            w_s[t * SW + sc + s] = acc + b2;
        }
        __syncthreads();
    }
    // memory scores (512 entries, hk_mem rows read once from global)
    for (int e = tid; e < 16 * TOPK; e += 128) {
        int t = e >> 5, k = e & 31;
        const float* hk  = hkm + ((size_t)(b * TT + t0 + t) * TOPK + k) * HID;
        const float* hqr = hq_s + t * HID;
        float acc = 0.f;
        for (int f = 0; f < HID; ++f) {
            float u = hqr[f] + hk[f];
            acc += (u > 0.f ? u : 0.f) * ws2_s[f];
        }
        w_s[t * SW + TT + k] = acc + b2;
    }
    __syncthreads();

    // softmax over 288 per t; also write f16 copy for the WMMA A fragment
    if (tid < 16) {
        float* r = w_s + tid * SW;
        _Float16* rh = wh_s + tid * 296;
        float mx = -3.4e38f;
        for (int i = 0; i < SW; ++i) mx = fmaxf(mx, r[i]);
        float sum = 0.f;
        for (int i = 0; i < SW; ++i) { float ev = expf(r[i] - mx); r[i] = ev; sum += ev; }
        float inv = 1.0f / sum;
        for (int i = 0; i < SW; ++i) { float wv = r[i] * inv; r[i] = wv; rh[i] = (_Float16)wv; }
    }
    __syncthreads();

    // out(16x64) = w_ctx(16x256) @ v(256x64) via WMMA; wave w owns cols [16w,16w+16)
    const int wave = tid >> 5, lane = tid & 31;
    const int g = lane >> 4, l = lane & 15;
    v8f acc = {};
    for (int k0 = 0; k0 < TT; k0 += 32) {
        // stage v tile 32x64 -> f16 transposed Vt_s[col][k]
        for (int i = tid; i < 512; i += 128) {
            int kr = i >> 4, c = (i & 15) * 4;
            float4 vv = *(const float4*)(vB + (size_t)(bh * TT + k0 + kr) * DH + c);
            Vt_s[(c + 0) * 40 + kr] = (_Float16)vv.x;
            Vt_s[(c + 1) * 40 + kr] = (_Float16)vv.y;
            Vt_s[(c + 2) * 40 + kr] = (_Float16)vv.z;
            Vt_s[(c + 3) * 40 + kr] = (_Float16)vv.w;
        }
        __syncthreads();

        const _Float16* wr = &wh_s[l * 296 + k0 + 8 * g];
        v8h alo = *(const v8h*)wr;          // ka = k0 + 8g + e      (e 0..7)
        v8h ahi = *(const v8h*)(wr + 16);   // ka = k0 + 16 + 8g + e (e 8..15)
        const _Float16* vr = &Vt_s[(wave * 16 + l) * 40 + 8 * g];
        v8h blo = *(const v8h*)vr;          // kb = 8g + e (e 0..7), 16 + 8g + e (e 8..15)
        v8h bhi = *(const v8h*)(vr + 16);
        v16h af, bf;
#pragma unroll
        for (int e = 0; e < 8; ++e) {
            af[e] = alo[e]; af[e + 8] = ahi[e];
            bf[e] = blo[e]; bf[e + 8] = bhi[e];
        }
        acc = __builtin_amdgcn_wmma_f32_16x16x32_f16(false, af, false, bf,
                                                     (short)0, acc, false, false);
        __syncthreads();
    }

    // memory-value contribution + gating + store
#pragma unroll
    for (int r = 0; r < 8; ++r) {
        int t = 8 * g + r;
        int d = wave * 16 + l;
        int rowt = b * TT + t0 + t;
        float o = acc[r];
        const float* wm = w_s + t * SW + TT;
        const float* vm = vmemA + (size_t)(rowt * TOPK) * DH + d;
        for (int k = 0; k < TOPK; ++k) o += wm[k] * vm[k * DH];
        o *= gates[rowt * HH + h];
        outF[rowt * (HH * DH) + h * DH + d] = o;
    }
}

// -----------------------------------------------------------------------------
extern "C" void kernel_launch(void* const* d_in, const int* in_sizes, int n_in,
                              void* d_out, int out_size, void* d_ws, size_t ws_size,
                              hipStream_t stream) {
    const float* x    = (const float*)d_in[0];
    const float* Wq   = (const float*)d_in[1];
    const float* Wk   = (const float*)d_in[2];
    const float* Wv   = (const float*)d_in[3];
    const float* Wo   = (const float*)d_in[4];
    const float* Wg1  = (const float*)d_in[5];
    const float* bg1  = (const float*)d_in[6];
    const float* Wg2  = (const float*)d_in[7];
    const float* bg2  = (const float*)d_in[8];
    const float* Ws1  = (const float*)d_in[9];
    const float* bs1  = (const float*)d_in[10];
    const float* Ws2  = (const float*)d_in[11];
    const float* bs2  = (const float*)d_in[12];
    const float* memk = (const float*)d_in[13];
    const float* memv = (const float*)d_in[14];
    (void)d_in[15]; (void)in_sizes; (void)n_in; (void)out_size; (void)ws_size;

    const int ROWS = BB * TT;          // 512
    const int BHT  = BB * HH * TT;     // 4096
    const int MROW = ROWS * TOPK;      // 16384

    char* w = (char*)d_ws;
    float* qB    = (float*)w; w += (size_t)BHT * DH * 4;
    float* kB    = (float*)w; w += (size_t)BHT * DH * 4;
    float* vB    = (float*)w; w += (size_t)BHT * DH * 4;
    float* g1    = (float*)w; w += (size_t)ROWS * GH * 4;
    float* gatesP= (float*)w; w += (size_t)ROWS * HH * 4;
    int*   idxP  = (int*)w;   w += (size_t)ROWS * TOPK * 4;
    float* kmemA = (float*)w; w += (size_t)MROW * DH * 4;
    float* vmemA = (float*)w; w += (size_t)MROW * DH * 4;
    float* hqP   = (float*)w; w += (size_t)BHT * HID * 4;
    float* hkcP  = (float*)w; w += (size_t)BHT * HID * 4;
    float* hkmP  = (float*)w; w += (size_t)MROW * HID * 4;
    float* outF  = (float*)w; w += (size_t)ROWS * HH * DH * 4;

    // q/k/v projections -> (B,H,T,DH)
    gemm_wmma_f16<<<dim3(DM/64, ROWS/64), 128, 0, stream>>>(x, Wq, nullptr, qB,
                                                            ROWS, DM, DM, 0, 1);
    gemm_wmma_f16<<<dim3(DM/64, ROWS/64), 128, 0, stream>>>(x, Wk, nullptr, kB,
                                                            ROWS, DM, DM, 0, 1);
    gemm_wmma_f16<<<dim3(DM/64, ROWS/64), 128, 0, stream>>>(x, Wv, nullptr, vB,
                                                            ROWS, DM, DM, 0, 1);
    // gate hidden layer (gelu)
    gemm_wmma_f16<<<dim3(GH/64, ROWS/64), 128, 0, stream>>>(x, Wg1, bg1, g1,
                                                            ROWS, GH, DM, 1, 0);
    gates_kernel<<<ROWS, 64, 0, stream>>>(g1, Wg2, bg2, gatesP);
    // memory selection + gather
    size_t smem = (size_t)(NMEM + DH + 256) * sizeof(float) + 256 * sizeof(int);
    memselect_kernel<<<ROWS, 256, smem, stream>>>(qB, memk, memv, idxP, kmemA, vmemA);
    // hidden projections (W1q = Ws1[:64], W1k = Ws1[64:])
    gemm_wmma_f16<<<dim3(HID/64, BHT/64), 128, 0, stream>>>(qB, Ws1, bs1, hqP,
                                                            BHT, HID, DH, 0, 0);
    gemm_wmma_f16<<<dim3(HID/64, BHT/64), 128, 0, stream>>>(kB, Ws1 + DH * HID, nullptr,
                                                            hkcP, BHT, HID, DH, 0, 0);
    gemm_wmma_f16<<<dim3(HID/64, MROW/64), 128, 0, stream>>>(kmemA, Ws1 + DH * HID, nullptr,
                                                             hkmP, MROW, HID, DH, 0, 0);
    // fused attention
    attn_kernel<<<dim3(TT/16, HH, BB), 128, 0, stream>>>(hqP, hkcP, hkmP, vB, vmemA,
                                                         gatesP, Ws2, bs2, outF);
    // output projection -> d_out
    gemm_wmma_f16<<<dim3(DM/64, ROWS/64), 128, 0, stream>>>(outF, Wo, nullptr,
                                                            (float*)d_out,
                                                            ROWS, DM, DM, 0, 0);
}
